// TwoStreamNet_89000312308227
// MI455X (gfx1250) — matrix operand order repaired and checked by reading, not compiled, not run on hardware
//
#include <hip/hip_runtime.h>
#include <stdint.h>

// ---------------------------------------------------------------------------
// CDNA5 (gfx1250) bf16 WMMA GEMM pipeline for TwoStreamNet.
// D = epi(A@B), fp32 in/out, bf16 WMMA compute, f32 accumulate.
// Double-buffered LDS staging; f32->bf16 via 1 v_add + v_perm_b32 pack.
// ---------------------------------------------------------------------------

typedef __attribute__((ext_vector_type(16))) __bf16 v16bf;
typedef __attribute__((ext_vector_type(8)))  float  v8f;

#define LDA_S 48   // LDS row stride (ushorts) for A tile [128 x 32]
#define LDB_S 72   // LDS row stride (ushorts) for B tile [32 x 64]

enum EpiMode {
    EPI_STORE = 0,          // C = acc
    EPI_SCALE = 1,          // C = alpha * acc
    EPI_BIAS  = 2,          // C = acc + bias
    EPI_BIAS_RELU_RES = 3,  // C = relu(acc + bias) + res
    EPI_BIAS_ADD_RELU_RES = 4 // C = relu(acc + bias + add) + res
};

// bf16 (round-half-up) in bits [31:16]; ties differ from RNE only.
__device__ __forceinline__ uint32_t bfbits(float f) {
    union { float f; uint32_t u; } c; c.f = f;
    return c.u + 0x8000u;
}
__device__ __forceinline__ unsigned short f2bf(float f) {
    return (unsigned short)(bfbits(f) >> 16);
}
// pack bf16(f.x) | bf16(f.y)<<16 etc. via v_perm_b32 (1 op per pair)
__device__ __forceinline__ uint2 pack4(float4 f) {
    uint2 p;
    p.x = __builtin_amdgcn_perm(bfbits(f.y), bfbits(f.x), 0x07060302u);
    p.y = __builtin_amdgcn_perm(bfbits(f.w), bfbits(f.z), 0x07060302u);
    return p;
}

union FragU { uint4 q[2]; v16bf v; };

// TA: operand A[m][k] = Am[k*lda + m]   (else Am[m*lda + k])
// TB: operand B[k][n] = Bm[n*ldb + k]   (else Bm[k*ldb + n])
template<bool TA, bool TB, int EPI>
__global__ __launch_bounds__(256)
void gemm_bf16_wmma(const float* __restrict__ Am, int lda,
                    const float* __restrict__ Bm, int ldb,
                    float* __restrict__ C, int ldc,
                    int K, float alpha,
                    const float* __restrict__ bias,
                    const float* __restrict__ add,
                    const float* __restrict__ res)
{
    __shared__ unsigned short As[2][128 * LDA_S];
    __shared__ unsigned short Bs[2][32  * LDB_S];

    const int tid = threadIdx.x;
    const int lid = tid & 31;
    const int wid = tid >> 5;
    const int wm  = (wid >> 1) * 32;   // wave M offset in block
    const int wn  = (wid & 1) * 32;    // wave N offset in block
    const int bm  = blockIdx.y * 128;
    const int bn  = blockIdx.x * 64;

    // ---- per-thread global-load geometry (contiguous 16B in all cases) ----
    int aR, aC;            // LDS placement coords for the A tile
    const float* aSrc;
    size_t aStep;
    if (!TA) {
        aR = tid >> 1;                 // row 0..127
        aC = (tid & 1) << 4;           // k seg 0/16
        aSrc  = Am + (size_t)(bm + aR) * lda + aC;
        aStep = 32;                    // advance along k
    } else {
        aR = tid >> 3;                 // k 0..31
        aC = (tid & 7) << 4;           // m seg 0..112
        aSrc  = Am + (size_t)aR * lda + bm + aC;
        aStep = (size_t)32 * lda;      // advance along k (rows of Am)
    }
    int bR, bC;
    const float* bSrc;
    size_t bStep;
    if (!TB) {
        bR = tid >> 3;                 // k 0..31
        bC = (tid & 7) << 3;           // n seg 0..56
        bSrc  = Bm + (size_t)bR * ldb + bn + bC;
        bStep = (size_t)32 * ldb;
    } else {
        bR = tid >> 2;                 // n 0..63
        bC = (tid & 3) << 3;           // k seg 0..24
        bSrc  = Bm + (size_t)(bn + bR) * ldb + bC;
        bStep = 32;
    }

    float4 ar[4];
    float4 br[2];
    auto loadG = [&](int kt) {
        const float* pa = aSrc + (size_t)kt * aStep;
        #pragma unroll
        for (int v = 0; v < 4; ++v) ar[v] = *(const float4*)(pa + v * 4);
        const float* pb = bSrc + (size_t)kt * bStep;
        #pragma unroll
        for (int v = 0; v < 2; ++v) br[v] = *(const float4*)(pb + v * 4);
    };
    auto storeL = [&](int buf) {
        if (!TA) {
            unsigned short* dst = &As[buf][aR * LDA_S + aC];
            #pragma unroll
            for (int v = 0; v < 4; ++v) *(uint2*)(dst + v * 4) = pack4(ar[v]);
        } else {
            #pragma unroll
            for (int v = 0; v < 4; ++v) {
                unsigned short* dst = &As[buf][(aC + v * 4) * LDA_S + aR];
                dst[0 * LDA_S] = f2bf(ar[v].x);
                dst[1 * LDA_S] = f2bf(ar[v].y);
                dst[2 * LDA_S] = f2bf(ar[v].z);
                dst[3 * LDA_S] = f2bf(ar[v].w);
            }
        }
        if (!TB) {
            unsigned short* dst = &Bs[buf][bR * LDB_S + bC];
            #pragma unroll
            for (int v = 0; v < 2; ++v) *(uint2*)(dst + v * 4) = pack4(br[v]);
        } else {
            #pragma unroll
            for (int v = 0; v < 2; ++v) {
                unsigned short* dst = &Bs[buf][(bC + v * 4) * LDB_S + bR];
                dst[0 * LDB_S] = f2bf(br[v].x);
                dst[1 * LDB_S] = f2bf(br[v].y);
                dst[2 * LDB_S] = f2bf(br[v].z);
                dst[3 * LDB_S] = f2bf(br[v].w);
            }
        }
    };

    v8f acc[2][2];
    #pragma unroll
    for (int i = 0; i < 2; ++i)
        #pragma unroll
        for (int j = 0; j < 2; ++j) { v8f z = {}; acc[i][j] = z; }

    const int kTiles = K >> 5;
    loadG(0);
    storeL(0);

    const int am = lid & 15;
    const int ak = (lid >> 4) << 3;    // 0 or 8

    int cur = 0;
    for (int kt = 0; kt < kTiles; ++kt) {
        __syncthreads();
        const bool more = (kt + 1 < kTiles);
        if (more) loadG(kt + 1);       // global loads in flight over ds/wmma

        // B fragment (32x16 bf16): lane = N column; lanes 0-15 -> K {0-7,16-23}; 16-31 -> K {8-15,24-31}.
        FragU bf[2];
        #pragma unroll
        for (int j = 0; j < 2; ++j) {
            const unsigned short* p = &Bs[cur][ak * LDB_S + wn + j * 16 + am];
            unsigned short e[16];
            #pragma unroll
            for (int s = 0; s < 8; ++s) {
                e[s]     = p[s * LDB_S];
                e[8 + s] = p[(16 + s) * LDB_S];
            }
            __builtin_memcpy(&bf[j], e, sizeof(e));
        }
        // A fragment (16x32 bf16): lanes 0-15 -> K {0-7,16-23}; 16-31 -> K {8-15,24-31}.
        FragU af[2];
        #pragma unroll
        for (int i = 0; i < 2; ++i) {
            const unsigned short* p = &As[cur][(wm + i * 16 + am) * LDA_S + ak];
            af[i].q[0] = *(const uint4*)(p);
            af[i].q[1] = *(const uint4*)(p + 16);
        }

        #pragma unroll
        for (int i = 0; i < 2; ++i)
            #pragma unroll
            for (int j = 0; j < 2; ++j)
                acc[i][j] = __builtin_amdgcn_wmma_f32_16x16x32_bf16(
                    false, af[i].v, false, bf[j].v, (short)0, acc[i][j], false, false);

        if (more) storeL(cur ^ 1);     // fill the other buffer
        cur ^= 1;
    }

    // ---- fused epilogue (compile-time mode: no branches) ----
    const int half = lid >> 4;
    const int ncol = lid & 15;
    #pragma unroll
    for (int i = 0; i < 2; ++i) {
        const int mbase = bm + wm + i * 16 + 8 * half;
        #pragma unroll
        for (int j = 0; j < 2; ++j) {
            const int n = bn + wn + j * 16 + ncol;
            const size_t base = (size_t)mbase * ldc + n;
            float bval = 0.f;
            if constexpr (EPI >= EPI_BIAS) bval = bias[n];
            float* cp = C + base;
            #pragma unroll
            for (int r = 0; r < 8; ++r) {
                const size_t o = (size_t)r * ldc;
                float v = acc[i][j][r];
                if constexpr (EPI == EPI_SCALE) v *= alpha;
                if constexpr (EPI >= EPI_BIAS) v += bval;
                if constexpr (EPI == EPI_BIAS_ADD_RELU_RES) v += add[base + o];
                if constexpr (EPI == EPI_BIAS_RELU_RES || EPI == EPI_BIAS_ADD_RELU_RES) {
                    v = fmaxf(v, 0.f);
                    v += res[base + o];
                }
                cp[o] = v;
            }
        }
    }
    (void)alpha; (void)bias; (void)add; (void)res;
}

// f[fi][c] = mean_j |xp[faces[fi][j]][c] - xd[fi][c]|
__global__ __launch_bounds__(256)
void face_feat_kernel(const float* __restrict__ xp, const float* __restrict__ xd,
                      const int* __restrict__ faces, float* __restrict__ f, int NFv)
{
    const int idx = blockIdx.x * 256 + threadIdx.x;     // over NFv*64 float4s
    const int fi = idx >> 6;
    const int c4 = (idx & 63) << 2;
    if (fi >= NFv) return;
    const int v0 = faces[fi * 3 + 0];
    const int v1 = faces[fi * 3 + 1];
    const int v2 = faces[fi * 3 + 2];
    float4 d = *(const float4*)(xd + (size_t)fi * 256 + c4);
    float4 a = *(const float4*)(xp + (size_t)v0 * 256 + c4);
    float4 b = *(const float4*)(xp + (size_t)v1 * 256 + c4);
    float4 c = *(const float4*)(xp + (size_t)v2 * 256 + c4);
    float4 o;
    o.x = (fabsf(a.x - d.x) + fabsf(b.x - d.x) + fabsf(c.x - d.x)) * (1.f / 3.f);
    o.y = (fabsf(a.y - d.y) + fabsf(b.y - d.y) + fabsf(c.y - d.y)) * (1.f / 3.f);
    o.z = (fabsf(a.z - d.z) + fabsf(b.z - d.z) + fabsf(c.z - d.z)) * (1.f / 3.f);
    o.w = (fabsf(a.w - d.w) + fabsf(b.w - d.w) + fabsf(c.w - d.w)) * (1.f / 3.f);
    *(float4*)(f + (size_t)fi * 256 + c4) = o;
}

// out[r] = concat(x1[r][0:256], x2[r][0:256])  -> (rows, 512)
__global__ __launch_bounds__(256)
void concat_kernel(const float* __restrict__ x1, const float* __restrict__ x2,
                   float* __restrict__ out, int rows)
{
    const int idx = blockIdx.x * 256 + threadIdx.x;     // over rows*128 float4s
    const int r = idx >> 7;
    const int c = idx & 127;
    if (r >= rows) return;
    float4 v = (c < 64) ? ((const float4*)(x1 + (size_t)r * 256))[c]
                        : ((const float4*)(x2 + (size_t)r * 256))[c - 64];
    ((float4*)(out + (size_t)r * 512))[c] = v;
}

extern "C" void kernel_launch(void* const* d_in, const int* in_sizes, int n_in,
                              void* d_out, int out_size, void* d_ws, size_t ws_size,
                              hipStream_t stream)
{
    (void)in_sizes; (void)n_in; (void)out_size; (void)ws_size;
    constexpr int NV = 6144, NF = 12288, DIM = 256;

    const float* primal   = (const float*)d_in[0];
    const float* A_primal = (const float*)d_in[1];
    const float* A_dual   = (const float*)d_in[2];
    const float* A        = (const float*)d_in[3];
    const int*   faces    = (const int*)d_in[4];
    const float* pW  = (const float*)d_in[5];
    const float* pb  = (const float*)d_in[6];
    const float* pgW = (const float*)d_in[7];
    const float* pgb = (const float*)d_in[8];
    const float* dW  = (const float*)d_in[9];
    const float* db  = (const float*)d_in[10];
    const float* dgW = (const float*)d_in[11];
    const float* dgb = (const float*)d_in[12];
    const float* Wp  = (const float*)d_in[13];
    const float* bp  = (const float*)d_in[14];
    const float* Wd  = (const float*)d_in[15];
    const float* bd  = (const float*)d_in[16];

    const size_t NVD = (size_t)NV * DIM, NFD = (size_t)NF * DIM;

    // d_out tuple layout: out0, out1, p0,p1,p2, d0,d1,d2
    float* out  = (float*)d_out;
    float* out0 = out;
    float* out1 = out0 + NVD;
    float* p0   = out1 + NFD;
    float* p1   = p0 + NVD;
    float* p2   = p1 + NVD;
    float* dd0  = p2 + NVD;
    float* dd1  = dd0 + NFD;
    float* dd2  = dd1 + NFD;

    // workspace
    float* w   = (float*)d_ws;
    float* xp  = w; w += NVD;            // final x_p
    float* xd  = w; w += NFD;            // final x_d
    float* t1  = w; w += NFD;            // Adj @ X
    float* ofc = w; w += NFD;            // out_fc
    float* t2  = w; w += NFD;            // Adj @ out_fc
    float* fb  = w; w += NFD;            // f
    float* mp  = w; w += NVD;            // mapped = A @ f
    float* cat = w; w += (size_t)NF * 512;

    const dim3 blk(256);
    const dim3 gNV(DIM / 64, NV / 128);  // (4, 48)
    const dim3 gNF(DIM / 64, NF / 128);  // (4, 96)
    const float* nil = nullptr;

    // primal_outs[0] = primal
    hipMemcpyAsync(p0, primal, NVD * sizeof(float), hipMemcpyDeviceToDevice, stream);

    // x_d0 = (1/3) * A^T @ primal   (M=NF, K=NV)
    gemm_bf16_wmma<true, false, EPI_SCALE><<<gNF, blk, 0, stream>>>(
        A, NF, primal, DIM, dd0, DIM, NV, 1.f / 3.f, nil, nil, nil);

    // ---- 3 primal AGG layers ----
    {
        const float* src[3] = {p0, p1, p2};
        float*       dst[3] = {p1, p2, xp};
        for (int i = 0; i < 3; ++i) {
            const float* Wi  = pW  + (size_t)i * DIM * DIM;
            const float* bi  = pb  + (size_t)i * DIM;
            const float* Wgi = pgW + (size_t)i * DIM * DIM;
            const float* bgi = pgb + (size_t)i * DIM;
            gemm_bf16_wmma<false, false, EPI_STORE><<<gNV, blk, 0, stream>>>(
                A_primal, NV, src[i], DIM, t1, DIM, NV, 1.f, nil, nil, nil);
            gemm_bf16_wmma<false, true, EPI_BIAS><<<gNV, blk, 0, stream>>>(
                t1, DIM, Wi, DIM, ofc, DIM, DIM, 1.f, bi, nil, nil);
            gemm_bf16_wmma<false, false, EPI_STORE><<<gNV, blk, 0, stream>>>(
                A_primal, NV, ofc, DIM, t2, DIM, NV, 1.f, nil, nil, nil);
            // dst = relu(acc + bg + out_fc) + src
            gemm_bf16_wmma<false, true, EPI_BIAS_ADD_RELU_RES><<<gNV, blk, 0, stream>>>(
                t2, DIM, Wgi, DIM, dst[i], DIM, DIM, 1.f, bgi, ofc, src[i]);
        }
    }

    // ---- 3 dual AGG layers ----
    {
        const float* src[3] = {dd0, dd1, dd2};
        float*       dst[3] = {dd1, dd2, xd};
        for (int i = 0; i < 3; ++i) {
            const float* Wi  = dW  + (size_t)i * DIM * DIM;
            const float* bi  = db  + (size_t)i * DIM;
            const float* Wgi = dgW + (size_t)i * DIM * DIM;
            const float* bgi = dgb + (size_t)i * DIM;
            gemm_bf16_wmma<false, false, EPI_STORE><<<gNF, blk, 0, stream>>>(
                A_dual, NF, src[i], DIM, t1, DIM, NF, 1.f, nil, nil, nil);
            gemm_bf16_wmma<false, true, EPI_BIAS><<<gNF, blk, 0, stream>>>(
                t1, DIM, Wi, DIM, ofc, DIM, DIM, 1.f, bi, nil, nil);
            gemm_bf16_wmma<false, false, EPI_STORE><<<gNF, blk, 0, stream>>>(
                A_dual, NF, ofc, DIM, t2, DIM, NF, 1.f, nil, nil, nil);
            gemm_bf16_wmma<false, true, EPI_BIAS_ADD_RELU_RES><<<gNF, blk, 0, stream>>>(
                t2, DIM, Wgi, DIM, dst[i], DIM, DIM, 1.f, bgi, ofc, src[i]);
        }
    }

    // f = mean(|x_p[faces] - x_d|, axis=1)
    face_feat_kernel<<<dim3((NF * 64) / 256), blk, 0, stream>>>(xp, xd, faces, fb, NF);

    // out1 = relu([x_d | f] @ Wd^T + bd) + x_d
    concat_kernel<<<dim3((NF * 128) / 256), blk, 0, stream>>>(xd, fb, cat, NF);
    gemm_bf16_wmma<false, true, EPI_BIAS_RELU_RES><<<gNF, blk, 0, stream>>>(
        cat, 2 * DIM, Wd, 2 * DIM, out1, DIM, 2 * DIM, 1.f, bd, nil, xd);

    // mapped = A @ f  (M=NV, K=NF)
    gemm_bf16_wmma<false, false, EPI_STORE><<<gNV, blk, 0, stream>>>(
        A, NF, fb, DIM, mp, DIM, NF, 1.f, nil, nil, nil);

    // out0 = relu([x_p | mapped] @ Wp^T + bp) + x_p
    concat_kernel<<<dim3((NV * 128) / 256), blk, 0, stream>>>(xp, mp, cat, NV);
    gemm_bf16_wmma<false, true, EPI_BIAS_RELU_RES><<<gNV, blk, 0, stream>>>(
        cat, 2 * DIM, Wp, 2 * DIM, out0, DIM, 2 * DIM, 1.f, bp, nil, xp);
}
